// CTG_EPA_Block_15719580303878
// MI455X (gfx1250) — matrix operand order repaired and checked by reading, not compiled, not run on hardware
//
#include <hip/hip_runtime.h>
#include <hip/hip_bf16.h>

typedef _Float16 half16 __attribute__((ext_vector_type(16)));
typedef _Float16 half8  __attribute__((ext_vector_type(8)));
typedef float    float8 __attribute__((ext_vector_type(8)));

#define DEVI static __device__ __forceinline__

constexpr int Bb = 4;
constexpr int C  = 128;
constexpr int Nn = 4096;                 // D*H*W = 16*16*16
constexpr int BN = Bb * Nn;              // 16384 tokens
constexpr float SC_CAUSAL = 0.08838834764831845f;  // 1/sqrt(C)
constexpr float SC_CH     = 0.015625f;             // 1/sqrt(N)

DEVI int laneid() { return threadIdx.x & 31; }
DEVI int waveid() { return threadIdx.x >> 5; }

DEVI half16 hcat(half8 lo, half8 hi) {
  return __builtin_shufflevector(lo, hi, 0,1,2,3,4,5,6,7,8,9,10,11,12,13,14,15);
}

// A fragment 16x32 f16: source row-major [m][k], row stride ld (halves).
// ISA layout: lanes 0-15 row m=lane, K {0..7,16..23}; lanes 16-31 K {8..15,24..31}.
DEVI half16 load_a(const _Float16* p, int ld) {
  const int l = laneid();
  const _Float16* r = p + (size_t)(l & 15) * ld + ((l >> 4) * 8);
  half8 lo = *reinterpret_cast<const half8*>(r);
  half8 hi = *reinterpret_cast<const half8*>(r + 16);
  return hcat(lo, hi);
}

// B fragment 32x16: B(k,n) = src[n*ld + k]  => computes A * src^T.
// lanes 0-15: n=lane, K {0..7,16..23}; lanes 16-31: K {8..15,24..31}.
DEVI half16 load_bt(const _Float16* p, int ld) {
  const int l = laneid();
  const _Float16* r = p + (size_t)(l & 15) * ld + ((l >> 4) * 8);
  half8 lo = *reinterpret_cast<const half8*>(r);
  half8 hi = *reinterpret_cast<const half8*>(r + 16);
  return hcat(lo, hi);
}

DEVI float8 wmma_f16(half16 a, half16 b, float8 c) {
  return __builtin_amdgcn_wmma_f32_16x16x32_f16(false, a, false, b, (short)0, c,
                                                false, false);
}

// reductions within a 16-lane half (rows of D fragment span 16 lanes)
DEVI float hmax16(float v) {
  v = fmaxf(v, __shfl_xor(v, 1));
  v = fmaxf(v, __shfl_xor(v, 2));
  v = fmaxf(v, __shfl_xor(v, 4));
  v = fmaxf(v, __shfl_xor(v, 8));
  return v;
}
DEVI float hsum16(float v) {
  v += __shfl_xor(v, 1);
  v += __shfl_xor(v, 2);
  v += __shfl_xor(v, 4);
  v += __shfl_xor(v, 8);
  return v;
}

// ---- CDNA5 async global->LDS copy (GLOBAL_LOAD_ASYNC_TO_LDS_B128, ASYNCcnt) ----
DEVI void cp_async16(_Float16* lds, const _Float16* g) {
  unsigned lofs = (unsigned)(unsigned long long)(uintptr_t)lds;  // wave LDS offset
  unsigned long long ga = (unsigned long long)(uintptr_t)g;
  asm volatile("global_load_async_to_lds_b128 %0, %1, off"
               :: "v"(lofs), "v"(ga) : "memory");
}
DEVI void wait_async0() {
#if defined(__has_builtin) && __has_builtin(__builtin_amdgcn_s_wait_asynccnt)
  __builtin_amdgcn_s_wait_asynccnt(0);
#else
  asm volatile("s_wait_asynccnt 0" ::: "memory");
#endif
}

// ---------------------------------------------------------------- LayerNorm
__global__ void ln_kernel(const float* __restrict__ x, const float* __restrict__ g,
                          const float* __restrict__ be, _Float16* __restrict__ xn) {
  const int b = blockIdx.x >> 5;
  const int n = ((blockIdx.x & 31) << 7) + threadIdx.x;  // 128 thr/block
  const float* xb = x + (size_t)b * C * Nn;
  float s = 0.f, ss = 0.f;
  for (int c = 0; c < C; ++c) {
    float v = xb[(size_t)c * Nn + n];
    s += v; ss += v * v;
  }
  const float mu = s * (1.f / C);
  const float var = ss * (1.f / C) - mu * mu;
  const float rs = rsqrtf(var + 1e-5f);
  _Float16* xo = xn + ((size_t)(b * Nn + n)) * C;
  for (int c = 0; c < C; ++c) {
    float v = xb[(size_t)c * Nn + n];
    xo[c] = (_Float16)((v - mu) * rs * g[c] + be[c]);
  }
}

// ------------------------------------------------------- pooled mean + gate
__global__ void pool_gate_kernel(const float* __restrict__ x,
                                 const float* __restrict__ Wg,
                                 const float* __restrict__ bg,
                                 float* __restrict__ gate) {
  __shared__ float pl[C];
  __shared__ float lg[3];
  const int b = blockIdx.x, c = threadIdx.x;
  const float* xb = x + ((size_t)b * C + c) * Nn;
  float s = 0.f;
  for (int n = 0; n < Nn; ++n) s += xb[n];
  pl[c] = s * (1.f / Nn);
  __syncthreads();
  if (c < 3) {
    float l = bg[c];
    for (int j = 0; j < C; ++j) l += pl[j] * Wg[c * C + j];
    lg[c] = l;
  }
  __syncthreads();
  if (c == 0) {
    float m = fmaxf(lg[0], fmaxf(lg[1], lg[2]));
    float e0 = __expf(lg[0] - m), e1 = __expf(lg[1] - m), e2 = __expf(lg[2] - m);
    float inv = 1.f / (e0 + e1 + e2);
    gate[b * 4 + 0] = e0 * inv;
    gate[b * 4 + 1] = e1 * inv;
    gate[b * 4 + 2] = e2 * inv;
  }
}

// -------------------------------------------------- fp32 -> f16 weight copy
__global__ void convw_kernel(const float* __restrict__ wq, const float* __restrict__ wk,
                             const float* __restrict__ wvc, const float* __restrict__ wvch,
                             _Float16* __restrict__ w16) {
  const int idx = blockIdx.x * 256 + threadIdx.x;     // 4*C*C = 65536
  const int which = idx >> 14, rem = idx & 16383;
  const float* src = which == 0 ? wq : which == 1 ? wk : which == 2 ? wvc : wvch;
  w16[idx] = (_Float16)src[rem];
}

// --------------------------------------------- projections (WMMA), q/k/vc/vch
__global__ void proj_kernel(const _Float16* __restrict__ xn,
                            const _Float16* __restrict__ w16,
                            const float* __restrict__ bq, const float* __restrict__ bk,
                            const float* __restrict__ bvc, const float* __restrict__ bvch,
                            _Float16* __restrict__ q16, _Float16* __restrict__ qT16,
                            _Float16* __restrict__ k16, _Float16* __restrict__ kT16,
                            _Float16* __restrict__ vcT16, _Float16* __restrict__ vch16) {
  const int widx = blockIdx.x * 8 + waveid();
  const int which = widx >> 13;            // 8192 tiles per matrix
  const int rem = widx & 8191;
  const int mt = rem >> 3, ot = rem & 7;
  const int m0 = mt * 16, o0 = ot * 16;
  const _Float16* W = w16 + (size_t)which * C * C;
  float8 acc = {};
#pragma unroll
  for (int kc = 0; kc < 4; ++kc)
    acc = wmma_f16(load_a(xn + (size_t)m0 * C + kc * 32, C),
                   load_bt(W + (size_t)o0 * C + kc * 32, C), acc);
  const int l = laneid(), col = l & 15, hi = l >> 4;
  const float* bp = which == 0 ? bq : which == 1 ? bk : which == 2 ? bvc : bvch;
  const float bv = bp[o0 + col];
  const int b = m0 >> 12, ml = m0 & (Nn - 1);
  _Float16* onrm = which == 0 ? q16 : which == 1 ? k16 : which == 3 ? vch16 : nullptr;
  _Float16* otr  = which == 0 ? qT16 : which == 1 ? kT16 : which == 2 ? vcT16 : nullptr;
#pragma unroll
  for (int i = 0; i < 8; ++i) {
    const float v = acc[i] + bv;
    const int row = m0 + i + 8 * hi;
    if (onrm) onrm[(size_t)row * C + o0 + col] = (_Float16)v;
    if (otr)  otr[((size_t)(b * C + o0 + col)) * Nn + (ml + i + 8 * hi)] = (_Float16)v;
  }
}

// ------------------------------------- causal branch: flash attention (WMMA)
// Block = 8 waves sharing one batch; K/V tiles staged in LDS via async copies.
__global__ void flash_kernel(const _Float16* __restrict__ q16,
                             const _Float16* __restrict__ k16,
                             const _Float16* __restrict__ vcT16,
                             _Float16* __restrict__ outc16) {
  __shared__ __align__(16) _Float16 ldsK[64][136];   // [mlocal][c], 68-dword rows
  __shared__ __align__(16) _Float16 ldsV[128][72];   // [c][mlocal], 36-dword rows
  __shared__ __align__(16) _Float16 ldsP[8][16][72]; // per-wave P staging
  const int wid = waveid();
  const int widx = blockIdx.x * 8 + wid;
  const int b = widx >> 8, mt = widx & 255;
  const int m0 = mt * 16;
  const int l = laneid(), col = l & 15, hi = l >> 4;

  const _Float16* qb = q16 + ((size_t)(b * Nn + m0)) * C;
  half16 aq[4];
#pragma unroll
  for (int kc = 0; kc < 4; ++kc) aq[kc] = load_a(qb + kc * 32, C);

  float8 acc[8] = {};
  float mr[8], lr[8];
#pragma unroll
  for (int i = 0; i < 8; ++i) { mr[i] = -3e38f; lr[i] = 0.f; }

  const _Float16* kb = k16 + (size_t)b * Nn * C;
  const _Float16* vb = vcT16 + (size_t)b * C * Nn;

  for (int jt = 0; jt < Nn / 64; ++jt) {
    const int n0 = jt * 64;
    __syncthreads();                       // previous tile fully consumed
    // cooperative async stage: K 64x128 (1024 x 16B), V 128x64 (1024 x 16B)
#pragma unroll
    for (int j = 0; j < 4; ++j) {
      const int idx = threadIdx.x + 256 * j;
      const int kr = idx >> 4, kc8 = idx & 15;
      cp_async16(&ldsK[kr][kc8 * 8], kb + (size_t)(n0 + kr) * C + kc8 * 8);
      const int vr = idx >> 3, vc8 = idx & 7;
      cp_async16(&ldsV[vr][vc8 * 8], vb + (size_t)vr * Nn + n0 + vc8 * 8);
    }
    wait_async0();
    __syncthreads();                       // tile visible to all waves
    if (jt + 1 < Nn / 64)
      __builtin_prefetch(kb + (size_t)(n0 + 64) * C, 0, 1);  // global_prefetch_b8

    float8 s[4] = {};
#pragma unroll
    for (int kc = 0; kc < 4; ++kc) {
#pragma unroll
      for (int st = 0; st < 4; ++st)
        s[st] = wmma_f16(aq[kc], load_bt(&ldsK[st * 16][kc * 32], 136), s[st]);
    }

#pragma unroll
    for (int i = 0; i < 8; ++i) {
      const float a0 = s[0][i] * SC_CAUSAL, a1 = s[1][i] * SC_CAUSAL;
      const float a2 = s[2][i] * SC_CAUSAL, a3 = s[3][i] * SC_CAUSAL;
      const float vmax = hmax16(fmaxf(fmaxf(a0, a1), fmaxf(a2, a3)));
      const float mnew = fmaxf(mr[i], vmax);
      const float alpha = __expf(mr[i] - mnew);
      const float p0 = __expf(a0 - mnew), p1 = __expf(a1 - mnew);
      const float p2 = __expf(a2 - mnew), p3 = __expf(a3 - mnew);
      lr[i] = lr[i] * alpha + hsum16(p0 + p1 + p2 + p3);
      mr[i] = mnew;
#pragma unroll
      for (int t = 0; t < 8; ++t) acc[t][i] *= alpha;
      _Float16* pr = &ldsP[wid][i + 8 * hi][col];
      pr[0]  = (_Float16)p0;
      pr[16] = (_Float16)p1;
      pr[32] = (_Float16)p2;
      pr[48] = (_Float16)p3;
    }
    const half16 ap0 = load_a(&ldsP[wid][0][0], 72);
    const half16 ap1 = load_a(&ldsP[wid][0][32], 72);
#pragma unroll
    for (int t = 0; t < 8; ++t) {
      acc[t] = wmma_f16(ap0, load_bt(&ldsV[t * 16][0], 72), acc[t]);
      acc[t] = wmma_f16(ap1, load_bt(&ldsV[t * 16][32], 72), acc[t]);
    }
  }

#pragma unroll
  for (int i = 0; i < 8; ++i) {
    const float inv = 1.f / lr[i];
    const int row = m0 + i + 8 * hi;
    _Float16* orow = outc16 + ((size_t)(b * Nn + row)) * C;
#pragma unroll
    for (int t = 0; t < 8; ++t) orow[t * 16 + col] = (_Float16)(acc[t][i] * inv);
  }
}

// ----------------------------------- channel branch: S = Q^T K (C x C, WMMA)
__global__ void chS_kernel(const _Float16* __restrict__ qT16,
                           const _Float16* __restrict__ kT16,
                           float* __restrict__ sch) {
  const int widx = blockIdx.x * 4 + waveid();
  const int b = widx >> 6, rem = widx & 63;
  const int ct = rem >> 3, dt = rem & 7;
  const _Float16* qa = qT16 + ((size_t)(b * C + ct * 16)) * Nn;
  const _Float16* kb = kT16 + ((size_t)(b * C + dt * 16)) * Nn;
  float8 acc = {};
  for (int n0 = 0; n0 < Nn; n0 += 32)
    acc = wmma_f16(load_a(qa + n0, Nn), load_bt(kb + n0, Nn), acc);
  const int l = laneid(), col = l & 15, hi = l >> 4;
#pragma unroll
  for (int i = 0; i < 8; ++i)
    sch[((size_t)(b * C + ct * 16 + i + 8 * hi)) * C + dt * 16 + col] = acc[i];
}

// channel softmax over d (rows of 128), output f16
__global__ void chsm_kernel(const float* __restrict__ sch, _Float16* __restrict__ attn16) {
  __shared__ float red[128];
  const int row = blockIdx.x, t = threadIdx.x;
  const float s = sch[(size_t)row * C + t] * SC_CH;
  red[t] = s; __syncthreads();
  for (int o = 64; o > 0; o >>= 1) { if (t < o) red[t] = fmaxf(red[t], red[t + o]); __syncthreads(); }
  const float m = red[0]; __syncthreads();
  const float e = __expf(s - m);
  red[t] = e; __syncthreads();
  for (int o = 64; o > 0; o >>= 1) { if (t < o) red[t] += red[t + o]; __syncthreads(); }
  attn16[(size_t)row * C + t] = (_Float16)(e / red[0]);
}

// out_channel = Vch (NxC) * Attn^T (CxC), WMMA
__global__ void chout_kernel(const _Float16* __restrict__ vch16,
                             const _Float16* __restrict__ attn16,
                             _Float16* __restrict__ outch16) {
  const int widx = blockIdx.x * 8 + waveid();
  const int b = widx >> 11, rem = widx & 2047;
  const int mt = rem >> 3, ct = rem & 7;
  const int m0 = mt * 16, c0 = ct * 16;
  float8 acc = {};
#pragma unroll
  for (int d0 = 0; d0 < C; d0 += 32)
    acc = wmma_f16(load_a(vch16 + ((size_t)(b * Nn + m0)) * C + d0, C),
                   load_bt(attn16 + ((size_t)(b * C + c0)) * C + d0, C), acc);
  const int l = laneid(), col = l & 15, hi = l >> 4;
#pragma unroll
  for (int i = 0; i < 8; ++i)
    outch16[((size_t)(b * Nn + m0 + i + 8 * hi)) * C + c0 + col] = (_Float16)acc[i];
}

// ------------------------------------------------------ Conv3d C->2, k=3 SAME
__global__ void conv_kernel(const _Float16* __restrict__ xn,
                            const float* __restrict__ Wc,
                            const float* __restrict__ bc,
                            float* __restrict__ cgf) {
  const int idx = blockIdx.x * 256 + threadIdx.x;  // B*2*N = 32768
  const int n = idx & (Nn - 1);
  const int o = (idx >> 12) & 1;
  const int b = idx >> 13;
  const int d = n >> 8, h = (n >> 4) & 15, w = n & 15;
  float acc = bc[o];
  for (int kd = 0; kd < 3; ++kd) {
    const int dd = d + kd - 1; if (dd < 0 || dd > 15) continue;
    for (int kh = 0; kh < 3; ++kh) {
      const int hh = h + kh - 1; if (hh < 0 || hh > 15) continue;
      for (int kw = 0; kw < 3; ++kw) {
        const int ww = w + kw - 1; if (ww < 0 || ww > 15) continue;
        const int nn = (dd << 8) | (hh << 4) | ww;
        const _Float16* xr = xn + ((size_t)(b * Nn + nn)) * C;
        const float* wr = Wc + (size_t)o * C * 27 + (kd * 9 + kh * 3 + kw);
        float a = 0.f;
        for (int c = 0; c < C; ++c) a += (float)xr[c] * wr[(size_t)c * 27];
        acc += a;
      }
    }
  }
  cgf[idx] = acc;
}

// --------------------------------------------------- gated fusion + residual
__global__ void fuse_kernel(const float* __restrict__ x,
                            const _Float16* __restrict__ outc16,
                            const _Float16* __restrict__ outch16,
                            const float* __restrict__ gate,
                            float* __restrict__ out) {
  const size_t idx = (size_t)blockIdx.x * 256 + threadIdx.x;  // B*C*N
  const int n = (int)(idx & (Nn - 1));
  const int c = (int)((idx >> 12) & (C - 1));
  const int b = (int)(idx >> 19);
  const float g0 = gate[b * 4 + 0], g2 = gate[b * 4 + 2];
  const size_t t = ((size_t)(b * Nn + n)) * C + c;
  out[idx] = x[idx] + g0 * (float)outc16[t] + g2 * (float)outch16[t];
}

extern "C" void kernel_launch(void* const* d_in, const int* in_sizes, int n_in,
                              void* d_out, int out_size, void* d_ws, size_t ws_size,
                              hipStream_t stream) {
  const float* x    = (const float*)d_in[0];
  const float* ln_g = (const float*)d_in[1];
  const float* ln_b = (const float*)d_in[2];
  const float* Wq   = (const float*)d_in[3];
  const float* bq   = (const float*)d_in[4];
  const float* Wk   = (const float*)d_in[5];
  const float* bk   = (const float*)d_in[6];
  const float* Wvc  = (const float*)d_in[7];
  const float* bvc  = (const float*)d_in[8];
  const float* Wvch = (const float*)d_in[9];
  const float* bvch = (const float*)d_in[10];
  const float* Wcv  = (const float*)d_in[11];
  const float* bcv  = (const float*)d_in[12];
  const float* Wg   = (const float*)d_in[13];
  const float* bg   = (const float*)d_in[14];

  float* out = (float*)d_out;                          // (B,C,D,H,W)
  float* cgf = out + (size_t)Bb * C * Nn;              // (B,2,D,H,W)

  char* ws = (char*)d_ws;
  const size_t SZH = (size_t)BN * C * sizeof(_Float16);   // 4 MiB
  _Float16* xn16    = (_Float16*)(ws + 0 * SZH);
  _Float16* q16     = (_Float16*)(ws + 1 * SZH);
  _Float16* qT16    = (_Float16*)(ws + 2 * SZH);
  _Float16* k16     = (_Float16*)(ws + 3 * SZH);
  _Float16* kT16    = (_Float16*)(ws + 4 * SZH);
  _Float16* vcT16   = (_Float16*)(ws + 5 * SZH);
  _Float16* vch16   = (_Float16*)(ws + 6 * SZH);
  _Float16* outc16  = (_Float16*)(ws + 7 * SZH);
  _Float16* outch16 = (_Float16*)(ws + 8 * SZH);
  char* tail = ws + 9 * SZH;
  _Float16* w16  = (_Float16*)tail;                 tail += (size_t)4 * C * C * 2;
  float*    sch  = (float*)tail;                    tail += (size_t)Bb * C * C * 4;
  _Float16* at16 = (_Float16*)tail;                 tail += (size_t)Bb * C * C * 2;
  float*    gate = (float*)tail;

  convw_kernel<<<256, 256, 0, stream>>>(Wq, Wk, Wvc, Wvch, w16);
  ln_kernel<<<Bb * (Nn / 128), 128, 0, stream>>>(x, ln_g, ln_b, xn16);
  pool_gate_kernel<<<Bb, 128, 0, stream>>>(x, Wg, bg, gate);
  proj_kernel<<<4096, 256, 0, stream>>>(xn16, w16, bq, bk, bvc, bvch,
                                        q16, qT16, k16, kT16, vcT16, vch16);
  flash_kernel<<<128, 256, 0, stream>>>(q16, k16, vcT16, outc16);
  chS_kernel<<<64, 128, 0, stream>>>(qT16, kT16, sch);
  chsm_kernel<<<Bb * C, 128, 0, stream>>>(sch, at16);
  chout_kernel<<<1024, 256, 0, stream>>>(vch16, at16, outch16);
  conv_kernel<<<(Bb * 2 * Nn) / 256, 256, 0, stream>>>(xn16, Wcv, bcv, cgf);
  fuse_kernel<<<((size_t)Bb * C * Nn) / 256, 256, 0, stream>>>(x, outc16, outch16, gate, out);
}
